// GatedGCNNet_1168231104594
// MI455X (gfx1250) — matrix operand order repaired and checked by reading, not compiled, not run on hardware
//
#include <hip/hip_runtime.h>
#include <hip/hip_bf16.h>

#define N_NODES 50000
#define N_EDGES 800000
#define D_IN    32
#define DD      96
#define NLAYERS 4
#define NC      10
#define EPS_F   1e-6f

typedef __attribute__((ext_vector_type(16))) __bf16 v16bf;
typedef __attribute__((ext_vector_type(8)))  __bf16 v8bf;
typedef __attribute__((ext_vector_type(8)))  float  v8f;

__device__ __forceinline__ unsigned short f2bf(float f) {
    __bf16 h = (__bf16)f;
    return __builtin_bit_cast(unsigned short, h);
}

// Stage W (KK x 96) into LDS pre-swizzled into per-lane B-fragment order:
// element i of lane ln in tile (ks,nt) = W[(ks*32 + (i>>3)*16 + (ln>>4)*8 + (i&7))*96 + nt*16 + (ln&15)]
template<int KSTEPS>
__device__ __forceinline__ void stage_wfrag(unsigned short* Wfrag, const float* __restrict__ W,
                                            int tid) {
    for (int f = tid; f < KSTEPS * 3072; f += 256) {
        int ks  = f / 3072;
        int rem = f - ks * 3072;
        int nt  = rem >> 9;
        int ln  = (rem >> 4) & 31;
        int i   = rem & 15;
        int k   = ks * 32 + ((i >> 3) << 4) + ((ln >> 4) << 3) + (i & 7);
        int col = nt * 16 + (ln & 15);
        Wfrag[f] = f2bf(W[k * DD + col]);
    }
}

// Core MMA: A tile rows from Xs (row-major bf16, stride KK), B from swizzled Wfrag.
template<int KSTEPS, int KK>
__device__ __forceinline__ void mma_tile(v8f acc[6], const unsigned short* Xs,
                                         const unsigned short* Wfrag,
                                         int rloc, int lh, int lane) {
    #pragma unroll
    for (int nt = 0; nt < 6; ++nt) {
        #pragma unroll
        for (int i = 0; i < 8; ++i) acc[nt][i] = 0.0f;
    }
    #pragma unroll
    for (int ks = 0; ks < KSTEPS; ++ks) {
        const v8bf a0 = *(const v8bf*)&Xs[rloc * KK + ks * 32 + lh * 8];
        const v8bf a1 = *(const v8bf*)&Xs[rloc * KK + ks * 32 + 16 + lh * 8];
        const v16bf a = __builtin_shufflevector(a0, a1,
            0, 1, 2, 3, 4, 5, 6, 7, 8, 9, 10, 11, 12, 13, 14, 15);
        #pragma unroll
        for (int nt = 0; nt < 6; ++nt) {
            const unsigned short* wp = &Wfrag[(((ks * 6 + nt) << 5) + lane) << 4];
            const v8bf b0 = *(const v8bf*)wp;
            const v8bf b1 = *(const v8bf*)(wp + 8);
            const v16bf b = __builtin_shufflevector(b0, b1,
                0, 1, 2, 3, 4, 5, 6, 7, 8, 9, 10, 11, 12, 13, 14, 15);
            acc[nt] = __builtin_amdgcn_wmma_f32_16x16x32_bf16(
                false, a, false, b, (short)0, acc[nt], false, false);
        }
    }
}

// Plain store epilogue with uniform full-tile fast path.
__device__ __forceinline__ void store_tile(const v8f acc[6], const float* bs,
                                           float* __restrict__ Y, int row0, int wave,
                                           int lh, int l16, int M) {
    const int rbase = row0 + wave * 16 + lh * 8;
    if (row0 + 128 <= M) {
        #pragma unroll
        for (int nt = 0; nt < 6; ++nt) {
            const int col = nt * 16 + l16;
            const float bcol = bs[col];
            #pragma unroll
            for (int r = 0; r < 8; ++r)
                Y[(size_t)(rbase + r) * DD + col] = acc[nt][r] + bcol;
        }
    } else {
        #pragma unroll
        for (int nt = 0; nt < 6; ++nt) {
            const int col = nt * 16 + l16;
            const float bcol = bs[col];
            #pragma unroll
            for (int r = 0; r < 8; ++r)
                if (rbase + r < M) Y[(size_t)(rbase + r) * DD + col] = acc[nt][r] + bcol;
        }
    }
}

// ---------------------------------------------------------------------------
// Generic [M x KK] @ [KK x 96] + bias GEMM via v_wmma_f32_16x16x32_bf16.
// Block = 256 threads (8 waves), tile = 128 rows x 96 cols, 16 rows per wave.
// FUSED==1: epilogue is the GatedGCN edge gate (Ce never hits HBM):
//   v = Dh[src] + Eh[dst] + Ce ; sigma = sigmoid(v)
//   num[dst] += Bh[src]*sigma ; den[dst] += sigma ; e += relu(v)
// ---------------------------------------------------------------------------
template<int FUSED, int KK>
__global__ __launch_bounds__(256)
void gemm_wmma_kernel(const float* __restrict__ X, const float* __restrict__ W,
                      const float* __restrict__ bias, float* __restrict__ Y,
                      int M,
                      const float* __restrict__ Dh, const float* __restrict__ Eh,
                      const float* __restrict__ Bh,
                      const int* __restrict__ src, const int* __restrict__ dst,
                      float* __restrict__ e_io, float* __restrict__ num,
                      float* __restrict__ den)
{
    constexpr int KSTEPS = KK / 32;
    __shared__ __align__(16) unsigned short Wfrag[KSTEPS * 6 * 32 * 16];
    __shared__ __align__(16) unsigned short Xs[128 * KK];
    __shared__ float bs[DD];

    const int tid  = threadIdx.x;
    const int wave = tid >> 5;
    const int lane = tid & 31;
    const int lh   = lane >> 4;
    const int l16  = lane & 15;
    const int row0 = blockIdx.x * 128;

    stage_wfrag<KSTEPS>(Wfrag, W, tid);
    for (int i = tid; i < DD; i += 256) bs[i] = bias[i];
    for (int i = tid; i < 128 * KK; i += 256) {
        int r = i / KK, k = i - r * KK;
        int row = row0 + r;
        float v = (row < M) ? X[(size_t)row * KK + k] : 0.0f;
        Xs[r * KK + k] = f2bf(v);
    }
    __syncthreads();

    v8f acc[6];
    mma_tile<KSTEPS, KK>(acc, Xs, Wfrag, wave * 16 + l16, lh, lane);

    if (FUSED == 0) {
        store_tile(acc, bs, Y, row0, wave, lh, l16, M);
    } else {
        const int rbase = row0 + wave * 16 + lh * 8;
        const bool full = (row0 + 128 <= M);
        #pragma unroll
        for (int r = 0; r < 8; ++r) {
            const int row = rbase + r;
            if (full || row < M) {
                const int s = src[row];
                const int t = dst[row];
                #pragma unroll
                for (int nt = 0; nt < 6; ++nt) {
                    const int col = nt * 16 + l16;
                    float ce = acc[nt][r] + bs[col];
                    float v  = Dh[(size_t)s * DD + col] + Eh[(size_t)t * DD + col] + ce;
                    float sg = 1.0f / (1.0f + __expf(-v));
                    unsafeAtomicAdd(&num[(size_t)t * DD + col], Bh[(size_t)s * DD + col] * sg);
                    unsafeAtomicAdd(&den[(size_t)t * DD + col], sg);
                    e_io[(size_t)row * DD + col] += fmaxf(v, 0.0f);
                }
            }
        }
    }
}

// ---------------------------------------------------------------------------
// Fused node-side kernel: computes Ah, Bh, Dh, Eh from one staged h tile.
// The X tile is staged once; the 18KB swizzled Wfrag is re-staged per weight
// between barriers (LDS stays at ~43KB).
// ---------------------------------------------------------------------------
__global__ __launch_bounds__(256)
void gemm4_wmma_kernel(const float* __restrict__ X,
                       const float* __restrict__ W0, const float* __restrict__ b0,
                       const float* __restrict__ W1, const float* __restrict__ b1,
                       const float* __restrict__ W2, const float* __restrict__ b2,
                       const float* __restrict__ W3, const float* __restrict__ b3,
                       float* __restrict__ Y0, float* __restrict__ Y1,
                       float* __restrict__ Y2, float* __restrict__ Y3,
                       int M)
{
    constexpr int KSTEPS = DD / 32;
    __shared__ __align__(16) unsigned short Wfrag[KSTEPS * 6 * 32 * 16];
    __shared__ __align__(16) unsigned short Xs[128 * DD];
    __shared__ float bs[DD];

    const int tid  = threadIdx.x;
    const int wave = tid >> 5;
    const int lane = tid & 31;
    const int lh   = lane >> 4;
    const int l16  = lane & 15;
    const int row0 = blockIdx.x * 128;

    const float* Wp[4] = {W0, W1, W2, W3};
    const float* bp[4] = {b0, b1, b2, b3};
    float*       Yp[4] = {Y0, Y1, Y2, Y3};

    for (int i = tid; i < 128 * DD; i += 256) {
        int r = i / DD, k = i - r * DD;
        int row = row0 + r;
        float v = (row < M) ? X[(size_t)row * DD + k] : 0.0f;
        Xs[r * DD + k] = f2bf(v);
    }

    for (int w = 0; w < 4; ++w) {
        stage_wfrag<KSTEPS>(Wfrag, Wp[w], tid);
        for (int i = tid; i < DD; i += 256) bs[i] = bp[w][i];
        __syncthreads();

        v8f acc[6];
        mma_tile<KSTEPS, DD>(acc, Xs, Wfrag, wave * 16 + l16, lh, lane);
        store_tile(acc, bs, Yp[w], row0, wave, lh, l16, M);

        __syncthreads();   // protect Wfrag/bs before next weight re-stage
    }
}

__global__ void fill_zero_kernel(float* __restrict__ p, int n) {
    int i = blockIdx.x * blockDim.x + threadIdx.x;
    for (; i < n; i += gridDim.x * blockDim.x) p[i] = 0.0f;
}

__global__ void node_update_kernel(float* __restrict__ h, const float* __restrict__ Ah,
                                   const float* __restrict__ num,
                                   const float* __restrict__ den, int n) {
    int i = blockIdx.x * blockDim.x + threadIdx.x;
    if (i < n) {
        float hn = Ah[i] + num[i] / (den[i] + EPS_F);
        h[i] += fmaxf(hn, 0.0f);
    }
}

// MLP readout 96 -> 48 -> 24 -> 10 with sigmoid; weights staged in LDS
__global__ __launch_bounds__(256)
void readout_kernel(const float* __restrict__ h,
                    const float* __restrict__ w0, const float* __restrict__ b0,
                    const float* __restrict__ w1, const float* __restrict__ b1,
                    const float* __restrict__ w2, const float* __restrict__ b2,
                    float* __restrict__ out, int n)
{
    __shared__ float W0[DD * 48]; __shared__ float B0[48];
    __shared__ float W1[48 * 24]; __shared__ float B1[24];
    __shared__ float W2[24 * NC]; __shared__ float B2[NC];
    const int tid = threadIdx.x;
    for (int i = tid; i < DD * 48; i += 256) W0[i] = w0[i];
    for (int i = tid; i < 48; i += 256)      B0[i] = b0[i];
    for (int i = tid; i < 48 * 24; i += 256) W1[i] = w1[i];
    for (int i = tid; i < 24; i += 256)      B1[i] = b1[i];
    for (int i = tid; i < 24 * NC; i += 256) W2[i] = w2[i];
    for (int i = tid; i < NC; i += 256)      B2[i] = b2[i];
    __syncthreads();

    const int node = blockIdx.x * 256 + tid;
    if (node >= n) return;

    float y0[48];
    #pragma unroll
    for (int j = 0; j < 48; ++j) y0[j] = B0[j];
    for (int k = 0; k < DD; ++k) {
        float hk = h[(size_t)node * DD + k];
        #pragma unroll
        for (int j = 0; j < 48; ++j) y0[j] = fmaf(hk, W0[k * 48 + j], y0[j]);
    }
    float y1[24];
    #pragma unroll
    for (int j = 0; j < 24; ++j) y1[j] = B1[j];
    #pragma unroll
    for (int k = 0; k < 48; ++k) {
        float a = fmaxf(y0[k], 0.0f);
        #pragma unroll
        for (int j = 0; j < 24; ++j) y1[j] = fmaf(a, W1[k * 24 + j], y1[j]);
    }
    float y2[NC];
    #pragma unroll
    for (int j = 0; j < NC; ++j) y2[j] = B2[j];
    #pragma unroll
    for (int k = 0; k < 24; ++k) {
        float a = fmaxf(y1[k], 0.0f);
        #pragma unroll
        for (int j = 0; j < NC; ++j) y2[j] = fmaf(a, W2[k * NC + j], y2[j]);
    }
    #pragma unroll
    for (int j = 0; j < NC; ++j)
        out[(size_t)node * NC + j] = 1.0f / (1.0f + __expf(-y2[j]));
}

extern "C" void kernel_launch(void* const* d_in, const int* in_sizes, int n_in,
                              void* d_out, int out_size, void* d_ws, size_t ws_size,
                              hipStream_t stream)
{
    const float* h_in = (const float*)d_in[0];
    const float* e_in = (const float*)d_in[1];
    const int*   src  = (const int*)d_in[2];
    const int*   dst  = (const int*)d_in[3];
    const float* fp_w = (const float*)d_in[4];
    const float* fp_b = (const float*)d_in[5];
    const float* ep_w = (const float*)d_in[6];
    const float* ep_b = (const float*)d_in[7];
    const float* Aw = (const float*)d_in[8];  const float* Ab = (const float*)d_in[9];
    const float* Bw = (const float*)d_in[10]; const float* Bb = (const float*)d_in[11];
    const float* Cw = (const float*)d_in[12]; const float* Cb = (const float*)d_in[13];
    const float* Dw = (const float*)d_in[14]; const float* Db = (const float*)d_in[15];
    const float* Ew = (const float*)d_in[16]; const float* Eb = (const float*)d_in[17];
    const float* m0w = (const float*)d_in[18]; const float* m0b = (const float*)d_in[19];
    const float* m1w = (const float*)d_in[20]; const float* m1b = (const float*)d_in[21];
    const float* m2w = (const float*)d_in[22]; const float* m2b = (const float*)d_in[23];

    const size_t ND = (size_t)N_NODES * DD;
    const size_t ED = (size_t)N_EDGES * DD;
    float* ws  = (float*)d_ws;
    float* h   = ws; ws += ND;
    float* e   = ws; ws += ED;
    float* Ah  = ws; ws += ND;
    float* Bh  = ws; ws += ND;
    float* Dh  = ws; ws += ND;
    float* Eh  = ws; ws += ND;
    float* num = ws; ws += ND;   // num & den adjacent -> one zero-fill
    float* den = ws; ws += ND;

    const dim3 blk(256);
    const int gN = (N_NODES + 127) / 128;
    const int gE = (N_EDGES + 127) / 128;

    // input projections (K = 32)
    gemm_wmma_kernel<0, D_IN><<<gN, blk, 0, stream>>>(h_in, fp_w, fp_b, h, N_NODES,
        nullptr, nullptr, nullptr, nullptr, nullptr, nullptr, nullptr, nullptr);
    gemm_wmma_kernel<0, D_IN><<<gE, blk, 0, stream>>>(e_in, ep_w, ep_b, e, N_EDGES,
        nullptr, nullptr, nullptr, nullptr, nullptr, nullptr, nullptr, nullptr);

    for (int l = 0; l < NLAYERS; ++l) {
        const size_t wo = (size_t)l * DD * DD;
        const size_t bo = (size_t)l * DD;

        // Ah/Bh/Dh/Eh from one staged h tile
        gemm4_wmma_kernel<<<gN, blk, 0, stream>>>(h,
            Aw + wo, Ab + bo, Bw + wo, Bb + bo, Dw + wo, Db + bo, Ew + wo, Eb + bo,
            Ah, Bh, Dh, Eh, N_NODES);

        fill_zero_kernel<<<4096, blk, 0, stream>>>(num, (int)(2 * ND));

        // fused edge GEMM + gate + scatter (updates e in place, fills num/den)
        gemm_wmma_kernel<1, DD><<<gE, blk, 0, stream>>>(e, Cw + wo, Cb + bo, nullptr, N_EDGES,
            Dh, Eh, Bh, src, dst, e, num, den);

        node_update_kernel<<<(int)((ND + 255) / 256), blk, 0, stream>>>(h, Ah, num, den, (int)ND);
    }

    readout_kernel<<<(N_NODES + 255) / 256, blk, 0, stream>>>(
        h, m0w, m0b, m1w, m1b, m2w, m2b, (float*)d_out, N_NODES);
}
